// FalconAttention_36197984371381
// MI455X (gfx1250) — matrix operand order repaired and checked by reading, not compiled, not run on hardware
//
#include <hip/hip_runtime.h>
#include <math.h>
#include <stdint.h>

// ---------------------------------------------------------------------------
// Falcon-7B attention block for MI455X (gfx1250, wave32, WMMA bf16).
//   - All three matmul stages on v_wmma_f32_16x16x32_bf16
//   - GEMM tiles double-buffered in LDS via global_load_async_to_lds_b128
//     (ASYNCcnt / s_wait_asynccnt), weights pre-transposed so both tiles are
//     K-contiguous and async-DMA friendly.
// ---------------------------------------------------------------------------

typedef __bf16 bf16_t;
typedef __attribute__((ext_vector_type(16))) __bf16 v16bf;
typedef __attribute__((ext_vector_type(8)))  float  v8f;

constexpr int kB    = 2;
constexpr int kS    = 1024;
constexpr int kH    = 4544;
constexpr int kNH   = 71;
constexpr int kHD   = 64;
constexpr int kNQKV = kH + 2 * kHD;   // 4672

// ---------------------------------------------------------------------------
// WMMA + async helpers
// ---------------------------------------------------------------------------

__device__ __forceinline__ v8f wmma_bf16(v16bf a, v16bf b, v8f c) {
  return __builtin_amdgcn_wmma_f32_16x16x32_bf16(
      /*neg_a=*/false, a, /*neg_b=*/false, b,
      /*c_mod=*/(short)0, c, /*reuse_a=*/false, /*reuse_b=*/false);
}

// One per-lane 16B async copy global -> LDS (GV mode, 64-bit vaddr).
__device__ __forceinline__ void async_copy_b128(uint32_t lds_addr,
                                                const void* gaddr) {
  asm volatile("global_load_async_to_lds_b128 %0, %1, off"
               :: "v"(lds_addr), "v"(gaddr) : "memory");
}
__device__ __forceinline__ void wait_async0() {
  asm volatile("s_wait_asynccnt 0x0" ::: "memory");
}
__device__ __forceinline__ uint32_t lds_off(const void* p) {
  return (uint32_t)(uintptr_t)p;   // LDS generic pointer: low 32 bits = offset
}

// A-matrix fragment (16x32 bf16): lane m=lane&15 owns row m; element i holds
// K = i + 16*half.
__device__ __forceinline__ v16bf frag_a(const bf16_t* row, int half) {
  v16bf f;
  const bf16_t* p1 = row + 16 * half;
#pragma unroll
  for (int i = 0; i < 16; ++i) f[i] = p1[i];
  return f;
}

// B-matrix fragment (32x16 bf16): lane n=lane&15 owns column n; lanes 0-15
// hold K=0..15, lanes 16-31 K=16..31. `col` points at K=0 of column n.
__device__ __forceinline__ v16bf frag_b(const bf16_t* col, int half) {
  v16bf f;
  const bf16_t* p = col + 16 * half;
#pragma unroll
  for (int i = 0; i < 16; ++i) f[i] = p[i];
  return f;
}

__device__ __forceinline__ float wave16_max(float v) {
#pragma unroll
  for (int m = 8; m >= 1; m >>= 1) v = fmaxf(v, __shfl_xor(v, m, 32));
  return v;
}
__device__ __forceinline__ float wave16_sum(float v) {
#pragma unroll
  for (int m = 8; m >= 1; m >>= 1) v += __shfl_xor(v, m, 32);
  return v;
}

// ---------------------------------------------------------------------------
// fp32 -> bf16 convert (plain)
// ---------------------------------------------------------------------------
__global__ void f32_to_bf16_kernel(const float* __restrict__ in,
                                   bf16_t* __restrict__ out, int n) {
  int i = blockIdx.x * blockDim.x + threadIdx.x;
  if (i < n) out[i] = (bf16_t)in[i];
}

// ---------------------------------------------------------------------------
// fp32 [K][N] -> bf16 [N][K] tiled transpose-convert (32x32 tiles via LDS)
// ---------------------------------------------------------------------------
__global__ __launch_bounds__(256) void transpose_f32_bf16_kernel(
    const float* __restrict__ in, bf16_t* __restrict__ out, int K, int N) {
  __shared__ float tile[32][33];
  const int kt = blockIdx.y * 32;
  const int nt = blockIdx.x * 32;
  const int tx = threadIdx.x;        // 0..31
  const int ty = threadIdx.y;        // 0..7
#pragma unroll
  for (int r = ty; r < 32; r += 8) {
    const int k = kt + r, n = nt + tx;
    tile[r][tx] = (k < K && n < N) ? in[(size_t)k * N + n] : 0.0f;
  }
  __syncthreads();
#pragma unroll
  for (int r = ty; r < 32; r += 8) {
    const int n = nt + r, k = kt + tx;
    if (n < N && k < K) out[(size_t)n * K + k] = (bf16_t)tile[tx][r];
  }
}

// ---------------------------------------------------------------------------
// Tiled WMMA GEMM: C[M,N](f32) = A[M,K](bf16,row) * Bt[N,K](bf16,row == B^T)
// 128x128x32 tile, 256 threads (8 waves 2x4), wave tile 64x32 (4x2 WMMA).
// Both tiles staged with async global->LDS b128 copies, double-buffered.
// M,K multiples of 128/32; N guarded per 128-row block (N % 16 == 0).
// ---------------------------------------------------------------------------
__global__ __launch_bounds__(256) void gemm_bf16_wmma(
    const bf16_t* __restrict__ A, const bf16_t* __restrict__ Bt,
    float* __restrict__ C, int M, int N, int K) {
  __shared__ __align__(16) bf16_t As[2][128][32];
  __shared__ __align__(16) bf16_t Bs[2][128][32];

  const int tid  = threadIdx.x;
  const int lane = tid & 31;
  const int wid  = tid >> 5;
  const int wm   = wid >> 2;       // 0..1
  const int wn   = wid & 3;        // 0..3
  const int half = lane >> 4;
  const int ln   = lane & 15;
  const int m0 = blockIdx.y * 128;
  const int n0 = blockIdx.x * 128;

  // tile staging mapping: each thread owns 32B (two b128 chunks) of each tile
  const int row = tid >> 1;            // 0..127
  const int ch  = (tid & 1) * 16;      // element offset 0 / 16
  const bool bvalid = (n0 + row) < N;  // uniform per wave (16 rows per wave)

  // zero invalid B rows once; async loads never touch them afterwards
  if (!bvalid) {
    const uint4 z = {0u, 0u, 0u, 0u};
    *reinterpret_cast<uint4*>(&Bs[0][row][ch])     = z;
    *reinterpret_cast<uint4*>(&Bs[0][row][ch + 8]) = z;
    *reinterpret_cast<uint4*>(&Bs[1][row][ch])     = z;
    *reinterpret_cast<uint4*>(&Bs[1][row][ch + 8]) = z;
  }

  v8f acc[4][2] = {};

  auto issue = [&](int buf, int k0) {
    const bf16_t* ga = A + (size_t)(m0 + row) * K + k0 + ch;
    async_copy_b128(lds_off(&As[buf][row][ch]), ga);
    async_copy_b128(lds_off(&As[buf][row][ch + 8]), ga + 8);
    if (bvalid) {
      const bf16_t* gb = Bt + (size_t)(n0 + row) * K + k0 + ch;
      async_copy_b128(lds_off(&Bs[buf][row][ch]), gb);
      async_copy_b128(lds_off(&Bs[buf][row][ch + 8]), gb + 8);
    }
  };

  issue(0, 0);
  wait_async0();
  __syncthreads();

  const int T = K / 32;
  for (int t = 0; t < T; ++t) {
    const int buf = t & 1;
    if (t + 1 < T) issue(buf ^ 1, (t + 1) * 32);   // prefetch next tile

    v16bf af[4], bfv[2];
#pragma unroll
    for (int mt = 0; mt < 4; ++mt)
      af[mt] = frag_a(&As[buf][wm * 64 + mt * 16 + ln][0], half);
#pragma unroll
    for (int nt = 0; nt < 2; ++nt)
      bfv[nt] = frag_b(&Bs[buf][wn * 32 + nt * 16 + ln][0], half);

#pragma unroll
    for (int mt = 0; mt < 4; ++mt)
#pragma unroll
      for (int nt = 0; nt < 2; ++nt)
        acc[mt][nt] = wmma_bf16(af[mt], bfv[nt], acc[mt][nt]);

    __syncthreads();   // all waves finished reading buf
    wait_async0();     // this wave's prefetch landed
    __syncthreads();   // everyone's prefetch landed
  }

  // epilogue: C/D layout rows m = j + 8*half, col n = lane&15
#pragma unroll
  for (int mt = 0; mt < 4; ++mt) {
#pragma unroll
    for (int nt = 0; nt < 2; ++nt) {
      const int c = n0 + wn * 32 + nt * 16 + ln;
      if (c < N) {
#pragma unroll
        for (int j = 0; j < 8; ++j) {
          const int r = m0 + wm * 64 + mt * 16 + j + 8 * half;
          C[(size_t)r * N + c] = acc[mt][nt][j];
        }
      }
    }
  }
}

// ---------------------------------------------------------------------------
// RoPE + head split. One thread per (b, s, head, i<32): dims i and i+32.
// Heads 0..70 -> Q [B,NH,S,HD]; 71 -> K [B,S,HD] (rotated); 72 -> V (plain).
// ---------------------------------------------------------------------------
__global__ void rope_split_kernel(const float* __restrict__ qkv,
                                  const int* __restrict__ pos,
                                  bf16_t* __restrict__ q, bf16_t* __restrict__ k,
                                  bf16_t* __restrict__ v) {
  int t = blockIdx.x * blockDim.x + threadIdx.x;
  const int total = kB * kS * 73 * 32;
  if (t >= total) return;
  const int i = t & 31;  t >>= 5;
  const int head = t % 73;  t /= 73;
  const int s = t % kS;
  const int b = t / kS;

  const float* src = qkv + (size_t)(b * kS + s) * kNQKV + head * kHD;
  const float x1 = src[i];
  const float x2 = src[i + 32];

  if (head == 72) {
    bf16_t* dst = v + (size_t)(b * kS + s) * kHD;
    dst[i] = (bf16_t)x1;
    dst[i + 32] = (bf16_t)x2;
    return;
  }
  const float p   = (float)pos[s];
  const float inv = __powf(10000.0f, -((float)(2 * i)) / 64.0f);
  float sn, c;
  __sincosf(p * inv, &sn, &c);
  const float o1 = x1 * c - x2 * sn;
  const float o2 = x2 * c + x1 * sn;
  bf16_t* dst = (head < kNH)
                    ? (q + (size_t)((b * kNH + head) * kS + s) * kHD)
                    : (k + (size_t)(b * kS + s) * kHD);
  dst[i] = (bf16_t)o1;
  dst[i + 32] = (bf16_t)o2;
}

// ---------------------------------------------------------------------------
// Flash attention, multi-query. One wave per (b, head, 16-query-row block).
// ---------------------------------------------------------------------------
__global__ __launch_bounds__(32) void flash_attn_kernel(
    const bf16_t* __restrict__ Q, const bf16_t* __restrict__ K,
    const bf16_t* __restrict__ V, bf16_t* __restrict__ O) {
  __shared__ __align__(16) bf16_t Pl[16][32];
  __shared__ __align__(16) bf16_t Vl[32][64];

  const int lane = threadIdx.x;
  const int half = lane >> 4;
  const int ln   = lane & 15;
  int blk = blockIdx.x;
  const int qb = blk % (kS / 16);  blk /= (kS / 16);
  const int h  = blk % kNH;
  const int b  = blk / kNH;
  const int q0 = qb * 16;

  const bf16_t* Qb = Q + (size_t)(b * kNH + h) * kS * kHD;
  const bf16_t* Kb = K + (size_t)b * kS * kHD;
  const bf16_t* Vb = V + (size_t)b * kS * kHD;

  v16bf qf0 = frag_a(Qb + (size_t)(q0 + ln) * kHD, half);
  v16bf qf1 = frag_a(Qb + (size_t)(q0 + ln) * kHD + 32, half);

  v8f o[4] = {};
  float mrow[8], lrow[8];
#pragma unroll
  for (int j = 0; j < 8; ++j) { mrow[j] = -3.0e38f; lrow[j] = 0.0f; }

  const int nkb = (q0 + 15) / 32 + 1;
  for (int kb = 0; kb < nkb; ++kb) {
    {
      const uint4* src =
          reinterpret_cast<const uint4*>(Vb + (size_t)(kb * 32 + lane) * kHD);
      uint4* dst = reinterpret_cast<uint4*>(&Vl[lane][0]);
#pragma unroll
      for (int i = 0; i < 8; ++i) dst[i] = src[i];
    }

    v8f s0 = {}, s1 = {};
    {
      const bf16_t* k0p = Kb + (size_t)(kb * 32 + ln) * kHD;
      const bf16_t* k1p = Kb + (size_t)(kb * 32 + 16 + ln) * kHD;
      s0 = wmma_bf16(qf0, frag_b(k0p, half), s0);
      s0 = wmma_bf16(qf1, frag_b(k0p + 32, half), s0);
      s1 = wmma_bf16(qf0, frag_b(k1p, half), s1);
      s1 = wmma_bf16(qf1, frag_b(k1p + 32, half), s1);
    }

#pragma unroll
    for (int j = 0; j < 8; ++j) {
      const int row = q0 + j + 8 * half;
      float s0j = s0[j] * 0.125f;
      float s1j = s1[j] * 0.125f;
      if (kb * 32 + ln > row)      s0j = -1.0e30f;
      if (kb * 32 + 16 + ln > row) s1j = -1.0e30f;
      const float mx   = wave16_max(fmaxf(s0j, s1j));
      const float mnew = fmaxf(mrow[j], mx);
      const float sc   = __expf(mrow[j] - mnew);
      const float p0   = __expf(s0j - mnew);
      const float p1   = __expf(s1j - mnew);
      lrow[j] = lrow[j] * sc + wave16_sum(p0 + p1);
      mrow[j] = mnew;
      o[0][j] *= sc; o[1][j] *= sc; o[2][j] *= sc; o[3][j] *= sc;
      Pl[j + 8 * half][ln]      = (bf16_t)p0;
      Pl[j + 8 * half][16 + ln] = (bf16_t)p1;
    }
    __syncthreads();

    v16bf pa = frag_a(&Pl[ln][0], half);
#pragma unroll
    for (int nb = 0; nb < 4; ++nb) {
      v16bf vf;
      const int d = nb * 16 + ln;
#pragma unroll
      for (int i = 0; i < 16; ++i) vf[i] = Vl[i + 16 * half][d];
      o[nb] = wmma_bf16(pa, vf, o[nb]);
    }
    __syncthreads();
  }

#pragma unroll
  for (int j = 0; j < 8; ++j) {
    const int row = q0 + j + 8 * half;
    const float inv_l = 1.0f / lrow[j];
#pragma unroll
    for (int nb = 0; nb < 4; ++nb) {
      const int d = nb * 16 + ln;
      O[(size_t)(b * kS + row) * kH + h * kHD + d] = (bf16_t)(o[nb][j] * inv_l);
    }
  }
}

// ---------------------------------------------------------------------------
// Launch
// ---------------------------------------------------------------------------
extern "C" void kernel_launch(void* const* d_in, const int* in_sizes, int n_in,
                              void* d_out, int out_size, void* d_ws,
                              size_t ws_size, hipStream_t stream) {
  (void)in_sizes; (void)n_in; (void)out_size; (void)ws_size;

  const float* hs   = (const float*)d_in[0];   // [B,S,H]
  const float* wqkv = (const float*)d_in[1];   // [H, H+2*HD]
  const float* wd   = (const float*)d_in[2];   // [H, H]
  // d_in[3]: attention_mask (pure causal -- computed analytically)
  const int* pos    = (const int*)d_in[4];     // [1,S]
  float* out        = (float*)d_out;           // [B,S,H] f32

  char* ws = (char*)d_ws;
  bf16_t* hs_b    = (bf16_t*)ws;  ws += (size_t)kB * kS * kH * 2;
  bf16_t* wqkv_t  = (bf16_t*)ws;  ws += (size_t)kH * kNQKV * 2;  // [NQKV][H]
  bf16_t* wd_t    = (bf16_t*)ws;  ws += (size_t)kH * kH * 2;     // [H][H] (T)
  float*  qkv     = (float*)ws;   ws += (size_t)kB * kS * kNQKV * 4;
  bf16_t* qbuf    = (bf16_t*)ws;  ws += (size_t)kB * kNH * kS * kHD * 2;
  bf16_t* kbuf    = (bf16_t*)ws;  ws += (size_t)kB * kS * kHD * 2;
  bf16_t* vbuf    = (bf16_t*)ws;  ws += (size_t)kB * kS * kHD * 2;
  bf16_t* attn_b  = (bf16_t*)ws;  // [B,S,NH*HD]

  const int n_hs = kB * kS * kH;
  f32_to_bf16_kernel<<<(n_hs + 255) / 256, 256, 0, stream>>>(hs, hs_b, n_hs);

  {  // W_qkv [H][NQKV] -> bf16 [NQKV][H]
    dim3 grid((kNQKV + 31) / 32, (kH + 31) / 32);
    transpose_f32_bf16_kernel<<<grid, dim3(32, 8), 0, stream>>>(wqkv, wqkv_t,
                                                                kH, kNQKV);
  }
  {  // W_dense [H][H] -> bf16 [H][H]^T
    dim3 grid((kH + 31) / 32, (kH + 31) / 32);
    transpose_f32_bf16_kernel<<<grid, dim3(32, 8), 0, stream>>>(wd, wd_t,
                                                                kH, kH);
  }

  // QKV projection: [2048, 4544] x [4544, 4672]
  {
    dim3 grid((kNQKV + 127) / 128, (kB * kS) / 128);
    gemm_bf16_wmma<<<grid, 256, 0, stream>>>(hs_b, wqkv_t, qkv,
                                             kB * kS, kNQKV, kH);
  }

  // RoPE + split into Q/K/V
  {
    const int n = kB * kS * 73 * 32;
    rope_split_kernel<<<(n + 255) / 256, 256, 0, stream>>>(qkv, pos, qbuf,
                                                           kbuf, vbuf);
  }

  // attention: one wave per (b, head, 16-row query block)
  flash_attn_kernel<<<kB * kNH * (kS / 16), 32, 0, stream>>>(qbuf, kbuf, vbuf,
                                                             attn_b);

  // dense projection: [2048, 4544] x [4544, 4544] -> d_out (f32)
  {
    dim3 grid((kH + 127) / 128, (kB * kS) / 128);
    gemm_bf16_wmma<<<grid, 256, 0, stream>>>(attn_b, wd_t, out,
                                             kB * kS, kH, kH);
  }
}
